// CopyRNN_44650480009331
// MI455X (gfx1250) — matrix operand order repaired and checked by reading, not compiled, not run on hardware
//
#include <hip/hip_runtime.h>

#define B_   64
#define L_   256
#define SH_  512
#define TH_  512
#define E_   256
#define V_   50000
#define OOV_ 100
#define VO_  (V_ + OOV_)

typedef float  v2f  __attribute__((ext_vector_type(2)));
typedef float  v8f  __attribute__((ext_vector_type(8)));
typedef __bf16 v16bf __attribute__((ext_vector_type(16)));

// ---------- WMMA tile helpers -------------------------------------------------
// D = A(16xK) * W^T(Kx16) + C ; A row-major (lda), W row-major (N x K, ldw)

__device__ __forceinline__ v8f wmma_f32_k4(const float* __restrict__ A, int lda,
                                           const float* __restrict__ W, int ldw,
                                           int m0, int n0, int k, v8f c) {
  int lane = threadIdx.x & 31;
  int half = lane >> 4, r = lane & 15;
  const float* ap = A + (long)(m0 + r) * lda + k + 2 * half;
  const float* bp = W + (long)(n0 + r) * ldw + k + 2 * half;
  v2f a, b;
  a.x = ap[0]; a.y = ap[1];
  b.x = bp[0]; b.y = bp[1];
  return __builtin_amdgcn_wmma_f32_16x16x4_f32(false, a, false, b, (short)0, c, false, false);
}

// A from bf16 pointer (global or LDS), B from bf16 pointer. Offsets pre-applied:
// ap -> A_row + k + 16*half ; bp -> W_row + k + 16*half
__device__ __forceinline__ v8f wmma_bf16_pp(const __bf16* ap, const __bf16* bp, v8f c) {
  v16bf a, b;
#pragma unroll
  for (int j = 0; j < 16; ++j) a[j] = ap[j];
#pragma unroll
  for (int j = 0; j < 16; ++j) b[j] = bp[j];
  return __builtin_amdgcn_wmma_f32_16x16x32_bf16(false, a, false, b, (short)0, c, false, false);
}

__device__ __forceinline__ void store_tile(float* D, int ldd, int m0, int n0, v8f c) {
  int lane = threadIdx.x & 31;
  int half = lane >> 4, r = lane & 15;
#pragma unroll
  for (int v = 0; v < 8; ++v)
    D[(long)(m0 + v + 8 * half) * ldd + n0 + r] = c[v];
}

__device__ __forceinline__ float sigf(float x) { return 1.0f / (1.0f + expf(-x)); }

// ---------- K1: copy_state (uniform masked mean) + x = [emb | copy_state] ----
__global__ void k_copystate_x(const int* __restrict__ prev_tok, const int* __restrict__ src,
                              const float* __restrict__ enc, const float* __restrict__ emb,
                              float* __restrict__ x) {
  int b = blockIdx.x, t = threadIdx.x;
  __shared__ int sflag[L_];
  __shared__ int sred[256];
  int tok = prev_tok[b];
  int m = (src[b * L_ + t] == tok) ? 1 : 0;
  sflag[t] = m; sred[t] = m;
  __syncthreads();
  for (int s = 128; s > 0; s >>= 1) { if (t < s) sred[t] += sred[t + s]; __syncthreads(); }
  int mc = sred[0];
  bool has = mc > 0;
  float inv = 1.0f / (float)(has ? (L_ - mc) : L_);
  for (int s = t; s < SH_; s += 256) {
    float acc = 0.f;
    for (int l = 0; l < L_; ++l)
      if (!(has && sflag[l])) acc += enc[((long)b * L_ + l) * SH_ + s];
    x[b * (E_ + SH_) + E_ + s] = acc * inv;
  }
  for (int e = t; e < E_; e += 256) x[b * (E_ + SH_) + e] = emb[(long)tok * E_ + e];
}

// ---------- K1b: W_copy -> bf16 ----------------------------------------------
__global__ void k_cvt_wcopy(const float* __restrict__ w, __bf16* __restrict__ wb) {
  int i = blockIdx.x * 256 + threadIdx.x;           // TH*SH = 262144
  wb[i] = (__bf16)w[i];
}

// ---------- K2: gates = x@W_ih^T + h@W_hh^T + b_ih + b_hh (fp32 WMMA) --------
__global__ void k_gates(const float* __restrict__ x, const float* __restrict__ Wih,
                        const float* __restrict__ h, const float* __restrict__ Whh,
                        const float* __restrict__ bih, const float* __restrict__ bhh,
                        float* __restrict__ gates) {
  int tile = blockIdx.x * 8 + (threadIdx.x >> 5);   // 512 tiles: 4 x 128
  int nt = tile & 127, mt = tile >> 7;
  int m0 = mt * 16, n0 = nt * 16;
  v8f c = {};
  for (int k = 0; k < E_ + SH_; k += 4) c = wmma_f32_k4(x, E_ + SH_, Wih, E_ + SH_, m0, n0, k, c);
  for (int k = 0; k < TH_; k += 4)      c = wmma_f32_k4(h, TH_, Whh, TH_, m0, n0, k, c);
  int r = threadIdx.x & 15;
  float bias = bih[n0 + r] + bhh[n0 + r];
#pragma unroll
  for (int v = 0; v < 8; ++v) c[v] += bias;
  store_tile(gates, 4 * TH_, m0, n0, c);
}

// ---------- K3: LSTM elementwise ---------------------------------------------
__global__ void k_lstm(const float* __restrict__ gates, const float* __restrict__ c_prev,
                       float* __restrict__ h_ws, float* __restrict__ cat2,
                       float* __restrict__ h_out, float* __restrict__ c_out) {
  int idx = blockIdx.x * 256 + threadIdx.x;         // B*TH = 32768
  int b = idx / TH_, j = idx % TH_;
  const float* g = gates + (long)b * 4 * TH_;
  float si = sigf(g[j]);
  float sf = sigf(g[TH_ + j]);
  float gg = tanhf(g[2 * TH_ + j]);
  float so = sigf(g[3 * TH_ + j]);
  float cn = sf * c_prev[idx] + si * gg;
  float hn = so * tanhf(cn);
  h_ws[idx] = hn;
  cat2[b * (SH_ + TH_) + SH_ + j] = hn;
  h_out[idx] = hn;
  c_out[idx] = cn;
}

// ---------- K4: q = h_new @ W_attn^T (fp32 WMMA) -----------------------------
__global__ void k_q(const float* __restrict__ h, const float* __restrict__ Wattn,
                    float* __restrict__ q) {
  int tile = blockIdx.x * 8 + (threadIdx.x >> 5);   // 128 tiles: 4 x 32
  int nt = tile & 31, mt = tile >> 5;
  v8f c = {};
  for (int k = 0; k < TH_; k += 4) c = wmma_f32_k4(h, TH_, Wattn, TH_, mt * 16, nt * 16, k, c);
  store_tile(q, SH_, mt * 16, nt * 16, c);
}

// ---------- K5: masked softmax attention + ctx into cat2 ---------------------
__global__ void k_attention(const float* __restrict__ q, const float* __restrict__ enc,
                            const int* __restrict__ pad, float* __restrict__ cat2) {
  int b = blockIdx.x, t = threadIdx.x;              // 256 threads, one per l
  __shared__ float sw[L_];
  __shared__ float sred[256];
  const float* er = enc + ((long)b * L_ + t) * SH_;
  const float* qb = q + (long)b * SH_;
  float acc = 0.f;
  for (int s = 0; s < SH_; ++s) acc += qb[s] * er[s];
  bool p = pad[b * L_ + t] != 0;
  float logit = p ? -__builtin_inff() : acc;
  sred[t] = logit; __syncthreads();
  for (int s = 128; s > 0; s >>= 1) { if (t < s) sred[t] = fmaxf(sred[t], sred[t + s]); __syncthreads(); }
  float mx = sred[0];
  __syncthreads();
  float e = p ? 0.f : expf(logit - mx);
  sw[t] = e; sred[t] = e; __syncthreads();
  for (int s = 128; s > 0; s >>= 1) { if (t < s) sred[t] += sred[t + s]; __syncthreads(); }
  float inv = 1.f / sred[0];
  for (int s = t; s < SH_; s += 256) {
    float cx = 0.f;
    for (int l = 0; l < L_; ++l) cx += sw[l] * enc[((long)b * L_ + l) * SH_ + s];
    cat2[b * (SH_ + TH_) + s] = cx * inv;
  }
}

// ---------- K6: attn_out = tanh([ctx|h] @ W_out^T + b_out) (fp32 WMMA) -------
// Also emits a bf16 copy of attn_out used as the A matrix of k_gen.
__global__ void k_attnout(const float* __restrict__ cat2, const float* __restrict__ Wout,
                          const float* __restrict__ bout, float* __restrict__ att_ws,
                          float* __restrict__ att_out, __bf16* __restrict__ att_bf) {
  int tile = blockIdx.x * 8 + (threadIdx.x >> 5);   // 128 tiles: 4 x 32
  int nt = tile & 31, mt = tile >> 5;
  v8f c = {};
  for (int k = 0; k < SH_ + TH_; k += 4)
    c = wmma_f32_k4(cat2, SH_ + TH_, Wout, SH_ + TH_, mt * 16, nt * 16, k, c);
  int lane = threadIdx.x & 31, half = lane >> 4, r = lane & 15;
  float bb = bout[nt * 16 + r];
#pragma unroll
  for (int v = 0; v < 8; ++v) {
    float val = tanhf(c[v] + bb);
    int m = mt * 16 + v + 8 * half, n = nt * 16 + r;
    att_ws[m * TH_ + n] = val;
    att_out[m * TH_ + n] = val;
    att_bf[m * TH_ + n] = (__bf16)val;
  }
}

// ---------- K7: total[:, :V] = exp(attn_out @ W_gen^T) (bf16 WMMA) -----------
// One wave per N-tile computes all 4 M-tiles: each W_gen element is loaded and
// converted exactly once (W_gen read = 102 MB = the HBM floor for this op).
__global__ void k_gen(const __bf16* __restrict__ att_bf, const float* __restrict__ Wgen,
                      float* __restrict__ total) {
  int nt = blockIdx.x * 8 + (threadIdx.x >> 5);     // 3125 N-tiles
  if (nt >= V_ / 16) return;
  int lane = threadIdx.x & 31, half = lane >> 4, r = lane & 15;
  int n0 = nt * 16;
  v8f c[4] = {{}, {}, {}, {}};
  for (int k = 0; k < TH_; k += 32) {
    const float* bp = Wgen + (long)(n0 + r) * TH_ + k + 16 * half;
    v16bf b;
#pragma unroll
    for (int j = 0; j < 16; ++j) b[j] = (__bf16)bp[j];
#pragma unroll
    for (int mt = 0; mt < 4; ++mt) {
      const __bf16* ap = att_bf + (long)(mt * 16 + r) * TH_ + k + 16 * half;
      v16bf a;
#pragma unroll
      for (int j = 0; j < 16; ++j) a[j] = ap[j];
      c[mt] = __builtin_amdgcn_wmma_f32_16x16x32_bf16(false, a, false, b, (short)0, c[mt],
                                                      false, false);
    }
  }
#pragma unroll
  for (int mt = 0; mt < 4; ++mt)
#pragma unroll
    for (int v = 0; v < 8; ++v) {
      int m = mt * 16 + v + 8 * half, n = n0 + r;
      total[(long)m * VO_ + n] = expf(c[mt][v]);
    }
}

// ---------- K8: OOV columns: exp(0) = 1 --------------------------------------
__global__ void k_oov(float* __restrict__ total) {
  int i = blockIdx.x * 256 + threadIdx.x;           // B*OOV = 6400
  if (i < B_ * OOV_) {
    int b = i / OOV_, j = i % OOV_;
    total[(long)b * VO_ + V_ + j] = 1.0f;
  }
}

// ---------- K9: copy scores (bf16 WMMA) + fused tanh/dot/exp/scatter ---------
// 4 waves/block; each wave stages its 16x512 enc tile into LDS as bf16 ONCE
// (16 KB/wave, 64 KB/block), then runs 32 N-tiles x 16 K-steps of WMMA reading
// A via ds_load and pre-converted bf16 W_copy from L2.
__global__ void k_copyscore(const float* __restrict__ enc, const __bf16* __restrict__ Wcb,
                            const float* __restrict__ attn, const int* __restrict__ pad,
                            const int* __restrict__ idx_oov, float* __restrict__ total) {
  __shared__ __bf16 sA[4 * 16 * SH_];               // 64 KB
  int w = threadIdx.x >> 5;
  int rb = blockIdx.x * 4 + w;                      // 1024 row-blocks of 16 enc rows
  int b = rb >> 4;
  int l0 = (rb & 15) * 16;
  int lane = threadIdx.x & 31, half = lane >> 4, r = lane & 15;
  __bf16* mySA = sA + w * 16 * SH_;

  // stage A tile: f32 -> bf16, coalesced; region is private to this wave and
  // same-wave LDS ops are in-order, so no workgroup barrier is needed.
  const float* arow = enc + (long)rb * 16 * SH_;
  for (int i = lane; i < 16 * SH_; i += 32) mySA[i] = (__bf16)arow[i];

  const float* attb = attn + (long)b * TH_;
  float acc[8] = {0.f, 0.f, 0.f, 0.f, 0.f, 0.f, 0.f, 0.f};
  for (int nt = 0; nt < 32; ++nt) {
    v8f c = {};
    for (int k = 0; k < SH_; k += 32) {
      const __bf16* ap = mySA + r * SH_ + k + 16 * half;
      const __bf16* bp = Wcb + (long)(nt * 16 + r) * SH_ + k + 16 * half;
      c = wmma_bf16_pp(ap, bp, c);
    }
    float av = attb[nt * 16 + r];
#pragma unroll
    for (int v = 0; v < 8; ++v) acc[v] += tanhf(c[v]) * av;
  }
#pragma unroll
  for (int v = 0; v < 8; ++v) {                     // reduce over 16 lanes per half
    float s = acc[v];
    s += __shfl_xor(s, 1, 32);
    s += __shfl_xor(s, 2, 32);
    s += __shfl_xor(s, 4, 32);
    s += __shfl_xor(s, 8, 32);
    acc[v] = s;
  }
  if (r == 0) {
#pragma unroll
    for (int v = 0; v < 8; ++v) {
      int l = l0 + 8 * half + v;
      bool p = pad[b * L_ + l] != 0;
      float e = p ? 0.f : expf(acc[v]);
      if (e != 0.f) atomicAdd(&total[(long)b * VO_ + idx_oov[b * L_ + l]], e);
    }
  }
}

// ---------- K10: total_prob = log(total / rowsum), in place ------------------
__global__ void k_norm(float* __restrict__ total) {
  int b = blockIdx.x, t = threadIdx.x;
  __shared__ float sred[256];
  float* row = total + (long)b * VO_;
  float s = 0.f;
  for (int v = t; v < VO_; v += 256) s += row[v];
  sred[t] = s; __syncthreads();
  for (int w = 128; w > 0; w >>= 1) { if (t < w) sred[t] += sred[t + w]; __syncthreads(); }
  float lg = logf(sred[0]);
  for (int v = t; v < VO_; v += 256) row[v] = logf(row[v]) - lg;
}

// ---------- launcher ---------------------------------------------------------
extern "C" void kernel_launch(void* const* d_in, const int* in_sizes, int n_in,
                              void* d_out, int out_size, void* d_ws, size_t ws_size,
                              hipStream_t stream) {
  const int*   prev_tok = (const int*)d_in[0];
  const float* enc      = (const float*)d_in[1];
  const int*   pad      = (const int*)d_in[2];
  // d_in[3] prev_context_state: provably unused (icp logits are {0,-inf})
  const float* prev_h   = (const float*)d_in[4];
  const float* prev_c   = (const float*)d_in[5];
  const int*   src      = (const int*)d_in[6];
  const int*   src_oov  = (const int*)d_in[7];
  // d_in[8] oov_counts: unused by reference
  const float* emb      = (const float*)d_in[9];
  const float* Wih      = (const float*)d_in[10];
  const float* Whh      = (const float*)d_in[11];
  const float* bih      = (const float*)d_in[12];
  const float* bhh      = (const float*)d_in[13];
  const float* Wattn    = (const float*)d_in[14];
  const float* Wout     = (const float*)d_in[15];
  const float* bout     = (const float*)d_in[16];
  const float* Wcopy    = (const float*)d_in[17];
  // d_in[18] W_icp: provably unused
  const float* Wgen     = (const float*)d_in[19];

  float* out   = (float*)d_out;
  float* total = out;                         // B x (V+OOV)
  float* att_o = out + (long)B_ * VO_;        // B x TH
  float* h_o   = att_o + (long)B_ * TH_;      // 1 x B x TH
  float* c_o   = h_o + (long)B_ * TH_;        // 1 x B x TH

  float* ws     = (float*)d_ws;
  float* x      = ws;                         // 64 x 768
  float* gates  = x + B_ * (E_ + SH_);        // 64 x 2048
  float* h_ws   = gates + B_ * 4 * TH_;       // 64 x 512
  float* q      = h_ws + B_ * TH_;            // 64 x 512
  float* cat2   = q + B_ * SH_;               // 64 x 1024
  float* att_ws = cat2 + B_ * (SH_ + TH_);    // 64 x 512
  __bf16* wcb    = (__bf16*)(att_ws + B_ * TH_); // 512 x 512 bf16
  __bf16* att_bf = wcb + TH_ * SH_;              // 64 x 512 bf16

  k_copystate_x<<<B_, 256, 0, stream>>>(prev_tok, src, enc, emb, x);
  k_cvt_wcopy<<<(TH_ * SH_) / 256, 256, 0, stream>>>(Wcopy, wcb);
  k_gates<<<64, 256, 0, stream>>>(x, Wih, prev_h, Whh, bih, bhh, gates);
  k_lstm<<<(B_ * TH_) / 256, 256, 0, stream>>>(gates, prev_c, h_ws, cat2, h_o, c_o);
  k_q<<<16, 256, 0, stream>>>(h_ws, Wattn, q);
  k_attention<<<B_, 256, 0, stream>>>(q, enc, pad, cat2);
  k_attnout<<<16, 256, 0, stream>>>(cat2, Wout, bout, att_ws, att_o, att_bf);
  k_gen<<<(V_ / 16 + 7) / 8, 256, 0, stream>>>(att_bf, Wgen, total);
  k_oov<<<(B_ * OOV_ + 255) / 256, 256, 0, stream>>>(total);
  k_copyscore<<<(B_ * L_ / 16) / 4, 128, 0, stream>>>(enc, wcb, att_ws, pad, src_oov, total);
  k_norm<<<B_, 256, 0, stream>>>(total);
}
